// BiMambaEncoderBlock_34102040330622
// MI455X (gfx1250) — compile-verified, matrix operand order checked by reading, not run on hardware
//
#include <hip/hip_runtime.h>
#include <hip/hip_bf16.h>
#include <math.h>

#define D_MODEL 256
#define D_INNER 512
#define D_STATE 64
#define D_CONV  4
#define DT_RANK 16
#define BATCH   2
#define SEQ     1024
#define NROWS   (BATCH * SEQ)            // 2048
#define XPROJ_N (DT_RANK + 2 * D_STATE)  // 144
#define LN_EPSF 1e-5f

typedef __attribute__((ext_vector_type(16))) __bf16 v16bf;
typedef __attribute__((ext_vector_type(8)))  __bf16 v8bf;
typedef __attribute__((ext_vector_type(8)))  float  v8f;

// ---------------------------------------------------------------------------
// Load one 16-bit WMMA fragment for this lane.  Per CDNA5 ISA
// §7.12.2, lane (row = lane&15, hi = lane>>4) holds K-chunks
// [k0+hi*8, +8) and [k0+16+hi*8, +8) of its row: two b128 loads.
// ---------------------------------------------------------------------------
__device__ inline v16bf load_frag16(const __bf16* p /* row base + k0 + hi*8 */) {
    const v8bf* q = reinterpret_cast<const v8bf*>(p);
    v8bf lo = q[0];   // K offsets +0..7
    v8bf hi = q[2];   // K offsets +16..23
    return __builtin_shufflevector(lo, hi, 0, 1, 2, 3, 4, 5, 6, 7,
                                           8, 9, 10, 11, 12, 13, 14, 15);
}

// ---------------------------------------------------------------------------
// WMMA GEMM:  C[M,N] = A[M,K] @ W[N,K]^T  (A, W bf16; C f32; +bias, +GELU)
// Block = 128 threads = 4 waves.  TWO=true: each wave computes a 16(M)x32(N)
// tile, one A fragment feeding two branch-free v_wmma_f32_16x16x32_bf16 per
// K-step.  TWO=false: single 16x16 tile (tail columns).
// M multiple of 64, K multiple of 32.  Tile columns start at n_base.
// ---------------------------------------------------------------------------
template <bool TWO>
__global__ void wmma_gemm_bf16(const __bf16* __restrict__ A,
                               const __bf16* __restrict__ W,
                               float* __restrict__ C,
                               int N, int K, int n_base,
                               const float* __restrict__ bias,
                               int act /*0=none, 1=exact gelu*/) {
    const int lane = threadIdx.x & 31;
    const int wave = threadIdx.x >> 5;
    const int row  = lane & 15;
    const int hi8  = (lane >> 4) << 3;

    const int m0 = ((blockIdx.x << 2) + wave) << 4;
    const int n0 = n_base + (blockIdx.y << (TWO ? 5 : 4));

    const __bf16* Arow = A + (size_t)(m0 + row) * K + hi8;
    const __bf16* W0   = W + (size_t)(n0 + row) * K + hi8;
    const __bf16* W1   = TWO ? (W + (size_t)(n0 + 16 + row) * K + hi8) : W0;

    v8f acc0 = {};
    v8f acc1 = {};

    for (int k0 = 0; k0 < K; k0 += 32) {
        if (k0 + 64 <= K) {  // gfx1250 global_prefetch_b8 for next K chunk
            __builtin_prefetch(Arow + k0 + 32, 0, 1);
            __builtin_prefetch(W0 + k0 + 32, 0, 1);
            if (TWO) __builtin_prefetch(W1 + k0 + 32, 0, 1);
        }
        const v16bf af  = load_frag16(Arow + k0);
        const v16bf bf0 = load_frag16(W0 + k0);
        acc0 = __builtin_amdgcn_wmma_f32_16x16x32_bf16(
            false, af, false, bf0, (short)0, acc0, false, false);
        if (TWO) {
            const v16bf bf1 = load_frag16(W1 + k0);
            acc1 = __builtin_amdgcn_wmma_f32_16x16x32_bf16(
                false, af, false, bf1, (short)0, acc1, false, false);
        }
    }

    // D layout: VGPR r -> M = r + 8*(lane>>4), N = lane&15
    const int n = n0 + row;
#pragma unroll
    for (int r = 0; r < 8; ++r) {
        const int m = m0 + r + hi8;
        float v = acc0[r];
        if (bias) v += bias[n];
        if (act == 1) v = 0.5f * v * (1.0f + erff(v * 0.70710678118654752f));
        C[(size_t)m * N + n] = v;
        if (TWO) {
            float u = acc1[r];
            if (bias) u += bias[n + 16];
            if (act == 1) u = 0.5f * u * (1.0f + erff(u * 0.70710678118654752f));
            C[(size_t)m * N + n + 16] = u;
        }
    }
}

// ---------------------------------------------------------------------------
// f32 -> bf16 conversion (for weights, once per GEMM).
// ---------------------------------------------------------------------------
__global__ void cvt_bf16_kernel(const float* __restrict__ src,
                                __bf16* __restrict__ dst, int n) {
    const int i = blockIdx.x * blockDim.x + threadIdx.x;
    if (i < n) dst[i] = (__bf16)src[i];
}

// ---------------------------------------------------------------------------
// LayerNorm over last dim (256); one block per row; optional seq-flip on read.
// Output bf16 (consumed only by the following WMMA GEMM).
// ---------------------------------------------------------------------------
__global__ void layernorm_kernel(const float* __restrict__ x,
                                 const float* __restrict__ g,
                                 const float* __restrict__ bta,
                                 __bf16* __restrict__ out, int flip) {
    __shared__ float red[D_MODEL];
    const int m   = blockIdx.x;
    const int tid = threadIdx.x;
    const int b   = m >> 10;
    const int t   = m & (SEQ - 1);
    const int ts  = flip ? (SEQ - 1 - t) : t;
    const float v = x[((size_t)(b * SEQ + ts)) * D_MODEL + tid];

    red[tid] = v;
    __syncthreads();
    for (int s = D_MODEL / 2; s > 0; s >>= 1) {
        if (tid < s) red[tid] += red[tid + s];
        __syncthreads();
    }
    const float mu = red[0] * (1.0f / D_MODEL);
    __syncthreads();
    const float d = v - mu;
    red[tid] = d * d;
    __syncthreads();
    for (int s = D_MODEL / 2; s > 0; s >>= 1) {
        if (tid < s) red[tid] += red[tid + s];
        __syncthreads();
    }
    const float var = red[0] * (1.0f / D_MODEL);
    out[(size_t)m * D_MODEL + tid] =
        (__bf16)(d * rsqrtf(var + LN_EPSF) * g[tid] + bta[tid]);
}

// ---------------------------------------------------------------------------
// Depthwise causal conv (width 4) over seq + SiLU.  xi = xz[:, :512].
// Writes f32 (scan/gate) and bf16 (xproj GEMM).
// ---------------------------------------------------------------------------
__global__ void conv_silu_kernel(const float* __restrict__ xz,
                                 const float* __restrict__ cw,
                                 const float* __restrict__ cb,
                                 float* __restrict__ xc,
                                 __bf16* __restrict__ xc_bf) {
    const int idx = blockIdx.x * blockDim.x + threadIdx.x; // [0, 2048*512)
    const int d = idx & (D_INNER - 1);
    const int m = idx >> 9;
    const int b = m >> 10;
    const int t = m & (SEQ - 1);
    float acc = cb[d];
#pragma unroll
    for (int k = 0; k < D_CONV; ++k) {
        const int ts = t - (D_CONV - 1) + k;
        if (ts >= 0)
            acc += cw[d * D_CONV + k] *
                   xz[((size_t)(b * SEQ + ts)) * (2 * D_INNER) + d];
    }
    const float s = acc / (1.0f + __expf(-acc));
    xc[(size_t)m * D_INNER + d] = s;
    xc_bf[(size_t)m * D_INNER + d] = (__bf16)s;
}

// ---------------------------------------------------------------------------
// dt = softplus(dbl[:, :16] @ dt_w^T + dt_b)  (K=16, plain VALU)
// ---------------------------------------------------------------------------
__global__ void dt_kernel(const float* __restrict__ dbl,
                          const float* __restrict__ dtw,
                          const float* __restrict__ dtb,
                          float* __restrict__ dt) {
    const int idx = blockIdx.x * blockDim.x + threadIdx.x; // [0, 2048*512)
    const int d = idx & (D_INNER - 1);
    const int m = idx >> 9;
    float acc = dtb[d];
    const float* row = dbl + (size_t)m * XPROJ_N;
#pragma unroll
    for (int r = 0; r < DT_RANK; ++r) acc += row[r] * dtw[d * DT_RANK + r];
    dt[(size_t)m * D_INNER + d] = (acc > 20.0f) ? acc : log1pf(__expf(acc));
}

// ---------------------------------------------------------------------------
// Selective scan.  One wave32 per (b, d); each lane holds states {lane, lane+32}.
// h_n <- exp(dt*A_n)*h_n + dt*B_n*xc ; y = sum_n h_n*C_n  (cross-lane reduce)
// ---------------------------------------------------------------------------
__global__ void scan_kernel(const float* __restrict__ dt,
                            const float* __restrict__ xc,
                            const float* __restrict__ dbl,
                            const float* __restrict__ A_log,
                            float* __restrict__ ys) {
    const int lane = threadIdx.x & 31;
    const int w    = threadIdx.x >> 5;
    const int p    = blockIdx.x * (blockDim.x >> 5) + w; // [0, 1024)
    const int b    = p >> 9;
    const int d    = p & (D_INNER - 1);

    const float A0 = -__expf(A_log[d * D_STATE + lane]);
    const float A1 = -__expf(A_log[d * D_STATE + lane + 32]);
    float h0 = 0.0f, h1 = 0.0f;

    for (int t = 0; t < SEQ; ++t) {
        const int m = b * SEQ + t;
        const float dtv = dt[(size_t)m * D_INNER + d];
        const float xcv = xc[(size_t)m * D_INNER + d];
        const float* row = dbl + (size_t)m * XPROJ_N;
        const float Bm0 = row[DT_RANK + lane];
        const float Bm1 = row[DT_RANK + lane + 32];
        const float Cm0 = row[DT_RANK + D_STATE + lane];
        const float Cm1 = row[DT_RANK + D_STATE + lane + 32];
        const float dbx = dtv * xcv;
        h0 = __expf(dtv * A0) * h0 + dbx * Bm0;
        h1 = __expf(dtv * A1) * h1 + dbx * Bm1;
        float y = h0 * Cm0 + h1 * Cm1;
#pragma unroll
        for (int off = 16; off >= 1; off >>= 1) y += __shfl_down(y, off);
        if (lane == 0) ys[(size_t)m * D_INNER + d] = y;
    }
}

// ---------------------------------------------------------------------------
// yb = (ys + xc*Dp) * silu(z),  z = xz[:, 512:]  -> bf16 for the out GEMM.
// ---------------------------------------------------------------------------
__global__ void gate_kernel(const float* __restrict__ ys,
                            const float* __restrict__ xc,
                            const float* __restrict__ xz,
                            const float* __restrict__ Dp,
                            __bf16* __restrict__ yb) {
    const int idx = blockIdx.x * blockDim.x + threadIdx.x;
    const int d = idx & (D_INNER - 1);
    const int m = idx >> 9;
    const float z = xz[(size_t)m * (2 * D_INNER) + D_INNER + d];
    const float y = ys[(size_t)m * D_INNER + d] +
                    xc[(size_t)m * D_INNER + d] * Dp[d];
    yb[(size_t)m * D_INNER + d] = (__bf16)(y * (z / (1.0f + __expf(-z))));
}

// ---------------------------------------------------------------------------
// x_ssm = out_f + flip(out_b); LayerNorm2 -> x2 (bf16 for final GEMM).
// ---------------------------------------------------------------------------
__global__ void add_flip_ln_kernel(const float* __restrict__ of,
                                   const float* __restrict__ ob,
                                   const float* __restrict__ g,
                                   const float* __restrict__ bta,
                                   __bf16* __restrict__ x2) {
    __shared__ float red[D_MODEL];
    const int m   = blockIdx.x;
    const int tid = threadIdx.x;
    const int b   = m >> 10;
    const int t   = m & (SEQ - 1);
    const int mb  = b * SEQ + (SEQ - 1 - t);
    const float v = of[(size_t)m * D_MODEL + tid] + ob[(size_t)mb * D_MODEL + tid];

    red[tid] = v;
    __syncthreads();
    for (int s = D_MODEL / 2; s > 0; s >>= 1) {
        if (tid < s) red[tid] += red[tid + s];
        __syncthreads();
    }
    const float mu = red[0] * (1.0f / D_MODEL);
    __syncthreads();
    const float d = v - mu;
    red[tid] = d * d;
    __syncthreads();
    for (int s = D_MODEL / 2; s > 0; s >>= 1) {
        if (tid < s) red[tid] += red[tid + s];
        __syncthreads();
    }
    const float var = red[0] * (1.0f / D_MODEL);
    x2[(size_t)m * D_MODEL + tid] =
        (__bf16)(d * rsqrtf(var + LN_EPSF) * g[tid] + bta[tid]);
}

// ---------------------------------------------------------------------------
extern "C" void kernel_launch(void* const* d_in, const int* in_sizes, int n_in,
                              void* d_out, int out_size, void* d_ws, size_t ws_size,
                              hipStream_t stream) {
    const float* x     = (const float*)d_in[0];
    const float* ln1_g = (const float*)d_in[19];
    const float* ln1_b = (const float*)d_in[20];
    const float* ln2_g = (const float*)d_in[21];
    const float* ln2_b = (const float*)d_in[22];
    const float* w2    = (const float*)d_in[23];
    const float* b2    = (const float*)d_in[24];

    // f32 scratch
    float* ws = (float*)d_ws;
    float* xz   = ws;                          // 2048*1024
    float* xc   = xz   + NROWS * 2 * D_INNER;  // 2048*512
    float* dbl  = xc   + NROWS * D_INNER;      // 2048*144
    float* dtb  = dbl  + NROWS * XPROJ_N;      // 2048*512
    float* ysf  = dtb  + NROWS * D_INNER;      // 2048*512
    float* outf = ysf  + NROWS * D_INNER;      // 2048*256
    float* outb = outf + NROWS * D_MODEL;      // 2048*256
    // bf16 scratch
    __bf16* bf   = (__bf16*)(outb + NROWS * D_MODEL);
    __bf16* xnb  = bf;                          // 2048*256
    __bf16* xcb  = xnb + NROWS * D_MODEL;       // 2048*512
    __bf16* ysb  = xcb + NROWS * D_INNER;       // 2048*512
    __bf16* x2b  = ysb + NROWS * D_INNER;       // 2048*256
    __bf16* wbf  = x2b + NROWS * D_MODEL;       // up to 1024*256 (weights)

    const int ew_blocks = (NROWS * D_INNER) / 256; // 4096

    for (int dir = 0; dir < 2; ++dir) {
        const int base = 1 + dir * 9;
        const float* in_w    = (const float*)d_in[base + 0];
        const float* conv_w  = (const float*)d_in[base + 1];
        const float* conv_b  = (const float*)d_in[base + 2];
        const float* xproj_w = (const float*)d_in[base + 3];
        const float* dt_w    = (const float*)d_in[base + 4];
        const float* dt_bias = (const float*)d_in[base + 5];
        const float* A_log   = (const float*)d_in[base + 6];
        const float* Dp      = (const float*)d_in[base + 7];
        const float* out_w   = (const float*)d_in[base + 8];
        float* out_dir = dir ? outb : outf;

        layernorm_kernel<<<NROWS, D_MODEL, 0, stream>>>(x, ln1_g, ln1_b, xnb, dir);

        cvt_bf16_kernel<<<(2 * D_INNER * D_MODEL) / 256, 256, 0, stream>>>(
            in_w, wbf, 2 * D_INNER * D_MODEL);
        wmma_gemm_bf16<true><<<dim3(NROWS / 64, (2 * D_INNER) / 32), 128, 0, stream>>>(
            xnb, wbf, xz, 2 * D_INNER, D_MODEL, 0, nullptr, 0);

        conv_silu_kernel<<<ew_blocks, 256, 0, stream>>>(xz, conv_w, conv_b, xc, xcb);

        cvt_bf16_kernel<<<(XPROJ_N * D_INNER) / 256, 256, 0, stream>>>(
            xproj_w, wbf, XPROJ_N * D_INNER);
        // N = 144: 128 columns via two-tile waves, 16-column tail single-tile.
        wmma_gemm_bf16<true><<<dim3(NROWS / 64, 4), 128, 0, stream>>>(
            xcb, wbf, dbl, XPROJ_N, D_INNER, 0, nullptr, 0);
        wmma_gemm_bf16<false><<<dim3(NROWS / 64, 1), 128, 0, stream>>>(
            xcb, wbf, dbl, XPROJ_N, D_INNER, 128, nullptr, 0);

        dt_kernel<<<ew_blocks, 256, 0, stream>>>(dbl, dt_w, dt_bias, dtb);

        scan_kernel<<<(BATCH * D_INNER) / 8, 256, 0, stream>>>(dtb, xc, dbl, A_log, ysf);

        gate_kernel<<<ew_blocks, 256, 0, stream>>>(ysf, xc, xz, Dp, ysb);

        cvt_bf16_kernel<<<(D_MODEL * D_INNER) / 256, 256, 0, stream>>>(
            out_w, wbf, D_MODEL * D_INNER);
        wmma_gemm_bf16<true><<<dim3(NROWS / 64, D_MODEL / 32), 128, 0, stream>>>(
            ysb, wbf, out_dir, D_MODEL, D_INNER, 0, nullptr, 0);
    }

    add_flip_ln_kernel<<<NROWS, D_MODEL, 0, stream>>>(outf, outb, ln2_g, ln2_b, x2b);

    cvt_bf16_kernel<<<(D_MODEL * D_MODEL) / 256, 256, 0, stream>>>(
        w2, wbf, D_MODEL * D_MODEL);
    wmma_gemm_bf16<true><<<dim3(NROWS / 64, D_MODEL / 32), 128, 0, stream>>>(
        x2b, wbf, (float*)d_out, D_MODEL, D_MODEL, 0, b2, 1);
}
